// GCNResnet18_6597069767374
// MI455X (gfx1250) — matrix operand order repaired and checked by reading, not compiled, not run on hardware
//
#include <hip/hip_runtime.h>
#include <hip/hip_bf16.h>

// ---------------------------------------------------------------------------
// Problem constants (from reference)
// ---------------------------------------------------------------------------
#define BATCH 32
#define G0    96
#define G1v   48
#define NPG0  (G0 * G0)          // 9216
#define NPG1  (G1v * G1v)        // 2304
#define N0    (BATCH * NPG0)     // 294912
#define N1    (BATCH * NPG1)     // 73728
#define NC    144
#define HID   512
#define LNEPS 1e-5f

typedef __attribute__((ext_vector_type(16))) __bf16 v16bf;
typedef __attribute__((ext_vector_type(8)))  float  v8f;

static __device__ __forceinline__ __bf16 f2bf(float x) { return (__bf16)x; }

// ---------------------------------------------------------------------------
// zero fill
// ---------------------------------------------------------------------------
__global__ void fill_zero_kernel(float* __restrict__ p, long n) {
    long i = (long)blockIdx.x * blockDim.x + threadIdx.x;
    long stride = (long)gridDim.x * blockDim.x;
    for (; i < n; i += stride) p[i] = 0.0f;
}

// fp32 -> bf16 elementwise (for the conv1 pooled activations)
__global__ void f32_to_bf16_kernel(const float* __restrict__ in,
                                   __bf16* __restrict__ out, long n) {
    long i = (long)blockIdx.x * blockDim.x + threadIdx.x;
    long stride = (long)gridDim.x * blockDim.x;
    for (; i < n; i += stride) out[i] = f2bf(in[i]);
}

// W[K,N] fp32 -> Wt[N,K] bf16 (per-layer weight convert+transpose)
__global__ void convT_bf16_kernel(const float* __restrict__ W,
                                  __bf16* __restrict__ Wt, int K, int N) {
    int t = blockIdx.x * blockDim.x + threadIdx.x;
    int total = K * N;
    if (t < total) {
        int k = t / N, n = t % N;
        Wt[(long)n * K + k] = f2bf(W[(long)k * N + n]);
    }
}

// ---------------------------------------------------------------------------
// Fused conv1 (3->64 GCN on 96x96 grid) + LayerNorm + ReLU + 2x2 max-pool.
// One 64-thread block per big-graph node; atomicMax(uint) into pooled [N1,64]
// (valid because ReLU output >= 0 and the buffer is pre-zeroed).
// ---------------------------------------------------------------------------
__global__ __launch_bounds__(64) void conv1_ln_pool_kernel(
    const float* __restrict__ x, const float* __restrict__ W,
    const float* __restrict__ bias, const float* __restrict__ gamma,
    const float* __restrict__ beta, float* __restrict__ pooled)
{
    const int node = blockIdx.x;
    const int c    = threadIdx.x;                 // 0..63
    const int b    = node / NPG0;
    const int p    = node - b * NPG0;
    const int i    = p / G0, j = p % G0;

    __shared__ float xs[5][3];                    // self + 4 neighbors
    const int  vi[5] = { i, i - 1, i + 1, i,     i     };
    const int  vj[5] = { j, j,     j,     j - 1, j + 1 };
    bool valid[5];
    #pragma unroll
    for (int q = 0; q < 5; ++q)
        valid[q] = (vi[q] >= 0) && (vi[q] < G0) && (vj[q] >= 0) && (vj[q] < G0);

    if (c < 15) {
        int q = c / 3, k = c % 3;
        int nn = b * NPG0 + vi[q] * G0 + vj[q];
        xs[q][k] = valid[q] ? x[(long)nn * 3 + k] : 0.0f;
    }
    __syncthreads();

    float di[5];
    #pragma unroll
    for (int q = 0; q < 5; ++q) {
        int cnt = (vi[q] > 0) + (vi[q] < G0 - 1) + (vj[q] > 0) + (vj[q] < G0 - 1);
        di[q] = rsqrtf((float)(cnt + 1));
    }

    const float w0 = W[0 * 64 + c], w1 = W[1 * 64 + c], w2 = W[2 * 64 + c];
    float z[5];
    #pragma unroll
    for (int q = 0; q < 5; ++q)
        z[q] = xs[q][0] * w0 + xs[q][1] * w1 + xs[q][2] * w2;

    float v = di[0] * di[0] * z[0] + bias[c];
    #pragma unroll
    for (int q = 1; q < 5; ++q)
        if (valid[q]) v += di[0] * di[q] * z[q];

    __shared__ float rs[64], rq[64];
    rs[c] = v; rq[c] = v * v;
    __syncthreads();
    for (int off = 32; off > 0; off >>= 1) {
        if (c < off) { rs[c] += rs[c + off]; rq[c] += rq[c + off]; }
        __syncthreads();
    }
    float mu  = rs[0] * (1.0f / 64.0f);
    float var = rq[0] * (1.0f / 64.0f) - mu * mu;
    float out = fmaxf((v - mu) * rsqrtf(var + LNEPS) * gamma[c] + beta[c], 0.0f);

    int c1 = b * NPG1 + (i >> 1) * G1v + (j >> 1);
    atomicMax((unsigned int*)&pooled[(long)c1 * 64 + c], __float_as_uint(out));
}

// ---------------------------------------------------------------------------
// bf16 WMMA GEMM:  Y[M,N](fp32) = X[M,K](bf16) @ W (bf16, pre-transposed [N,K])
// Tile BM=128, BN=64, BK=32. 256 threads = 8 waves; wave w owns rows 16w..16w+15
// and 4 N-fragments of 16. Staging = one 16B copy per thread per tile,
// register double-buffered so global loads overlap WMMA on the previous tile.
// ---------------------------------------------------------------------------
#define BM 128
#define BN 64
#define BK 32

__global__ __launch_bounds__(256) void gemm_bf16_wmma_kernel(
    const __bf16* __restrict__ X, const __bf16* __restrict__ Wt,
    float* __restrict__ Y, int M, int K, int N)
{
    __shared__ __bf16 As[BM][BK];   // 8 KB
    __shared__ __bf16 Bs[BN][BK];   // 4 KB ([n][k]: K contiguous)

    const int tid  = threadIdx.x;
    const int wave = tid >> 5;
    const int lane = tid & 31;
    const int half = lane >> 4;
    const int l16  = lane & 15;

    const int rowBase = blockIdx.y * BM;
    const int colBase = blockIdx.x * BN;

    // 16-byte staging assignment per thread
    const int ar = tid >> 1, ak = (tid & 1) * 16;   // A: 128 rows x 2 segs
    const int br = tid >> 2, bk = (tid & 3) * 8;    // B: 64 rows x 4 segs

    const __bf16* Asrc = &X[(long)(rowBase + ar) * K + ak];
    const __bf16* Bsrc = &Wt[(long)(colBase + br) * K + bk];

    ulonglong2 ra = *reinterpret_cast<const ulonglong2*>(Asrc);
    ulonglong2 rb = *reinterpret_cast<const ulonglong2*>(Bsrc);

    v8f acc[4] = {};

    for (int kt = 0; kt < K; kt += BK) {
        *reinterpret_cast<ulonglong2*>(&As[ar][ak]) = ra;
        *reinterpret_cast<ulonglong2*>(&Bs[br][bk]) = rb;
        __syncthreads();

        if (kt + BK < K) {  // prefetch next tile into registers (overlaps WMMA)
            ra = *reinterpret_cast<const ulonglong2*>(Asrc + kt + BK);
            rb = *reinterpret_cast<const ulonglong2*>(Bsrc + kt + BK);
        }

        // A fragment: 16x32 bf16, M = l16 for all lanes, half selects K-phase.
        v16bf a;
        #pragma unroll
        for (int iv = 0; iv < 8; ++iv) {
            int k0 = (iv & 3) * 2 + half * 8 + (iv >> 2) * 16;
            a[2 * iv]     = As[wave * 16 + l16][k0];
            a[2 * iv + 1] = As[wave * 16 + l16][k0 + 1];
        }
        #pragma unroll
        for (int f = 0; f < 4; ++f) {
            // B fragment: 32x16 bf16. N = l16, K = half*8 + (e&7) + (e>>3)*16.
            v16bf bfrag;
            #pragma unroll
            for (int e = 0; e < 16; ++e)
                bfrag[e] = Bs[f * 16 + l16][half * 8 + (e & 7) + (e >> 3) * 16];
            acc[f] = __builtin_amdgcn_wmma_f32_16x16x32_bf16(
                false, a, false, bfrag, (short)0, acc[f], false, false);
        }
        __syncthreads();
    }

    // C layout: VGPR r -> M = r + 8*half, N = l16
    #pragma unroll
    for (int f = 0; f < 4; ++f)
        #pragma unroll
        for (int r = 0; r < 8; ++r) {
            int m = rowBase + wave * 16 + half * 8 + r;
            int n = colBase + f * 16 + l16;
            Y[(long)m * N + n] = acc[f][r];
        }
}

// ---------------------------------------------------------------------------
// Per-node epilogue on the 48x48 pooled grid. Mode flags:
//   F_AGG : GCN aggregation (analytic grid gather) + bias
//   F_LN  : LayerNorm(gamma,beta) + ReLU
//   F_RES : add residual row
// Writes fp32 (if out) and/or a bf16 mirror (if outb) for the next GEMM.
// ---------------------------------------------------------------------------
#define F_AGG 1
#define F_LN  2
#define F_RES 4

__global__ __launch_bounds__(128) void node_kernel(
    const float* __restrict__ Y, const float* __restrict__ bias,
    const float* __restrict__ gamma, const float* __restrict__ beta,
    const float* __restrict__ resid, float* __restrict__ out,
    __bf16* __restrict__ outb, int fo, int mode)
{
    const int node = blockIdx.x;
    const int tid  = threadIdx.x;
    const int p    = node % NPG1;
    const int base = node - p;
    const int i    = p / G1v, j = p % G1v;

    int   nb[4];
    float dnb[4];
    int   nnb = 0;
    float ns  = 0.0f;
    if (mode & F_AGG) {
        int cnt = (i > 0) + (i < G1v - 1) + (j > 0) + (j < G1v - 1);
        float dself = rsqrtf((float)(cnt + 1));
        ns = dself * dself;
        const int ni[4] = { i - 1, i + 1, i,     i     };
        const int nj[4] = { j,     j,     j - 1, j + 1 };
        #pragma unroll
        for (int q = 0; q < 4; ++q) {
            if (ni[q] >= 0 && ni[q] < G1v && nj[q] >= 0 && nj[q] < G1v) {
                int c2 = (ni[q] > 0) + (ni[q] < G1v - 1) + (nj[q] > 0) + (nj[q] < G1v - 1);
                nb[nnb]  = base + ni[q] * G1v + nj[q];
                dnb[nnb] = rsqrtf((float)(c2 + 1)) * dself;
                ++nnb;
            }
        }
    }

    float z[4];
    float sum = 0.0f, sumsq = 0.0f;
    int s = 0;
    for (int c = tid; c < fo; c += 128, ++s) {
        float v;
        if (mode & F_AGG) {
            v = ns * Y[(long)node * fo + c] + bias[c];
            for (int q = 0; q < nnb; ++q)
                v += dnb[q] * Y[(long)nb[q] * fo + c];
        } else {
            v = Y[(long)node * fo + c];
        }
        z[s] = v; sum += v; sumsq += v * v;
    }

    if (mode & F_LN) {
        __shared__ float rs[128], rq[128];
        rs[tid] = sum; rq[tid] = sumsq;
        __syncthreads();
        for (int off = 64; off > 0; off >>= 1) {
            if (tid < off) { rs[tid] += rs[tid + off]; rq[tid] += rq[tid + off]; }
            __syncthreads();
        }
        float mu  = rs[0] / (float)fo;
        float var = rq[0] / (float)fo - mu * mu;
        float inv = rsqrtf(var + LNEPS);
        s = 0;
        for (int c = tid; c < fo; c += 128, ++s) {
            float v = fmaxf((z[s] - mu) * inv * gamma[c] + beta[c], 0.0f);
            if (out)  out[(long)node * fo + c]  = v;
            if (outb) outb[(long)node * fo + c] = f2bf(v);
        }
    } else {
        s = 0;
        for (int c = tid; c < fo; c += 128, ++s) {
            float v = z[s];
            if (mode & F_RES) v += resid[(long)node * fo + c];
            if (out)  out[(long)node * fo + c]  = v;
            if (outb) outb[(long)node * fo + c] = f2bf(v);
        }
    }
}

// ---------------------------------------------------------------------------
// 4x4 avg-pool over the pooled grid into [B*NC, 512] (analytic gather).
// ---------------------------------------------------------------------------
__global__ __launch_bounds__(256) void avgpool_kernel(
    const float* __restrict__ h, float* __restrict__ hp)
{
    const int cl  = blockIdx.x;          // 0 .. B*NC-1
    const int b   = cl / NC;
    const int cc  = cl % NC;
    const int ci  = cc / 12, cj = cc % 12;
    for (int c = threadIdx.x; c < HID; c += 256) {
        float sum = 0.0f;
        #pragma unroll
        for (int r = 0; r < 4; ++r)
            #pragma unroll
            for (int t = 0; t < 4; ++t) {
                int n = b * NPG1 + (4 * ci + r) * G1v + (4 * cj + t);
                sum += h[(long)n * HID + c];
            }
        hp[(long)cl * HID + c] = sum * (1.0f / 16.0f);
    }
}

// ---------------------------------------------------------------------------
// FC: out[b,o] = hp(flat)[b, :] . fcW[:, o] + fcb[o]   (K = NC*HID = 73728)
// ---------------------------------------------------------------------------
__global__ __launch_bounds__(256) void fc_kernel(
    const float* __restrict__ hp, const float* __restrict__ fcW,
    const float* __restrict__ fcb, float* __restrict__ out)
{
    const int bo = blockIdx.x;
    const int b  = bo / NC, o = bo % NC;
    const long Kfc = (long)NC * HID;
    float acc = 0.0f;
    for (long k = threadIdx.x; k < Kfc; k += 256)
        acc += hp[(long)b * Kfc + k] * fcW[k * NC + o];
    __shared__ float rs[256];
    rs[threadIdx.x] = acc;
    __syncthreads();
    for (int off = 128; off > 0; off >>= 1) {
        if (threadIdx.x < off) rs[threadIdx.x] += rs[threadIdx.x + off];
        __syncthreads();
    }
    if (threadIdx.x == 0) out[(long)b * NC + o] = rs[0] + fcb[o];
}

// ---------------------------------------------------------------------------
// Launch sequence
// ---------------------------------------------------------------------------
extern "C" void kernel_launch(void* const* d_in, const int* in_sizes, int n_in,
                              void* d_out, int out_size, void* d_ws, size_t ws_size,
                              hipStream_t stream)
{
    // Input order: x; params in insertion order (conv1_W, conv1_b, bn1_g, bn1_b,
    // blocks[0..15]{W1,b1,g1,t1,W2,b2,g2,t2}, fc_W, fc_b); index arrays unused
    // (grid/cluster structure is analytic).
    const float* x      = (const float*)d_in[0];
    const float* conv1W = (const float*)d_in[1];
    const float* conv1b = (const float*)d_in[2];
    const float* bn1g   = (const float*)d_in[3];
    const float* bn1b   = (const float*)d_in[4];
    const int    NBLK   = 16;
    const float* fcW    = (const float*)d_in[5 + NBLK * 8 + 0];
    const float* fcb    = (const float*)d_in[5 + NBLK * 8 + 1];

    // workspace layout
    float*  bufH  = (float*)d_ws;                       // fp32 activation (for avgpool / conv1 pool)
    float*  bufY  = bufH + (size_t)N1 * HID;            // GEMM output scratch
    float*  bufC  = bufY + (size_t)N1 * HID;            // h1 (residual source)
    float*  hp    = bufC + (size_t)N1 * HID;            // [B*NC, 512]
    __bf16* bufHb = (__bf16*)(hp + (size_t)BATCH * NC * HID);  // bf16 GEMM input mirror
    __bf16* bufWt = bufHb + (size_t)N1 * HID;           // bf16 transposed weights (<=512x512)

    // stage 1: zero pooled region, fused conv1+LN+ReLU+maxpool, bf16 mirror
    {
        long n = (long)N1 * 64;
        fill_zero_kernel<<<dim3(1024), dim3(256), 0, stream>>>(bufH, n);
        conv1_ln_pool_kernel<<<dim3(N0), dim3(64), 0, stream>>>(
            x, conv1W, conv1b, bn1g, bn1b, bufH);
        f32_to_bf16_kernel<<<dim3(1024), dim3(256), 0, stream>>>(bufH, bufHb, n);
    }

    static const int FI[16] = {64,64,64, 64, 128,128,128, 128,
                               256,256,256,256,256, 256, 512,512};
    static const int FO[16] = {64,64,64, 128,128,128,128, 256,
                               256,256,256,256,256, 512, 512,512};

    for (int blk = 0; blk < NBLK; ++blk) {
        const int pb = 5 + blk * 8;
        const float* W1 = (const float*)d_in[pb + 0];
        const float* b1 = (const float*)d_in[pb + 1];
        const float* g1 = (const float*)d_in[pb + 2];
        const float* t1 = (const float*)d_in[pb + 3];
        const float* W2 = (const float*)d_in[pb + 4];
        const float* b2 = (const float*)d_in[pb + 5];
        const float* g2 = (const float*)d_in[pb + 6];
        const float* t2 = (const float*)d_in[pb + 7];
        const int fi = FI[blk], fo = FO[blk];

        // t = h @ W1
        convT_bf16_kernel<<<dim3((fi * fo + 255) / 256), dim3(256), 0, stream>>>(
            W1, bufWt, fi, fo);
        gemm_bf16_wmma_kernel<<<dim3(fo / BN, N1 / BM), dim3(256), 0, stream>>>(
            bufHb, bufWt, bufY, N1, fi, fo);
        // h1 = relu(ln(agg(t) + t*ns + b1))  (fp32 only; residual source)
        node_kernel<<<dim3(N1), dim3(128), 0, stream>>>(
            bufY, b1, g1, t1, nullptr, bufC, nullptr, fo, F_AGG | F_LN);
        // u = relu(ln(h1; g2, t2))  (bf16 only; next GEMM input)
        node_kernel<<<dim3(N1), dim3(128), 0, stream>>>(
            bufC, nullptr, g2, t2, nullptr, nullptr, bufHb, fo, F_LN);
        // y2 = u @ W2
        convT_bf16_kernel<<<dim3((fo * fo + 255) / 256), dim3(256), 0, stream>>>(
            W2, bufWt, fo, fo);
        gemm_bf16_wmma_kernel<<<dim3(fo / BN, N1 / BM), dim3(256), 0, stream>>>(
            bufHb, bufWt, bufY, N1, fo, fo);
        // h = h1 + (agg(y2) + y2*ns + b2)  (fp32 for avgpool + bf16 for next block)
        node_kernel<<<dim3(N1), dim3(128), 0, stream>>>(
            bufY, b2, nullptr, nullptr, bufC, bufH, bufHb, fo, F_AGG | F_RES);
    }

    // stage 3: 4x4 avg pool + FC
    avgpool_kernel<<<dim3(BATCH * NC), dim3(256), 0, stream>>>(bufH, hp);
    fc_kernel<<<dim3(BATCH * NC), dim3(256), 0, stream>>>(hp, fcW, fcb, (float*)d_out);
}
